// Linear_90950227460263
// MI455X (gfx1250) — matrix operand order repaired and checked by reading, not compiled, not run on hardware
//
#include <hip/hip_runtime.h>
#include <hip/hip_bf16.h>

typedef __attribute__((ext_vector_type(16))) __bf16 v16bf;
typedef __attribute__((ext_vector_type(8)))  float  v8f;
typedef __attribute__((ext_vector_type(4)))  float  v4f;

#define IN_F   7168
#define OUT_F  18432
#define TOK    64
#define BLK    128          // scale block size
#define NKB    (IN_F/BLK)   // 56 K-blocks

// convert 16 f32 (4x v4f) to a v16bf fragment (lowers to v_cvt_pk_bf16_f32)
__device__ __forceinline__ v16bf cvt16(v4f a, v4f b, v4f c, v4f d) {
  v16bf r;
#pragma unroll
  for (int i = 0; i < 4; ++i) {
    r[i]      = (__bf16)a[i];
    r[4 + i]  = (__bf16)b[i];
    r[8 + i]  = (__bf16)c[i];
    r[12 + i] = (__bf16)d[i];
  }
  return r;
}

// load a 16-contiguous-f32 B fragment (per-lane K-contiguous along a W row)
__device__ __forceinline__ v16bf loadB(const float* p) {
  v4f b0 = *(const v4f*)(p);
  v4f b1 = *(const v4f*)(p + 4);
  v4f b2 = *(const v4f*)(p + 8);
  v4f b3 = *(const v4f*)(p + 12);
  return cvt16(b0, b1, b2, b3);
}

// One workgroup = 4 waves = 128 threads.
// Wave w owns M-tile w (tokens 16w..16w+15) and computes TWO N-tiles
// (n0 = blockIdx*32, n1 = n0+16) sharing one A fragment per WMMA step.
// Block covers N = 32 output features, all 64 tokens -> weight_q streamed
// from DRAM exactly once; x stays L2-resident.
__global__ __launch_bounds__(128)
void dequant_gemm_bf16wmma(const float* __restrict__ x,
                           const float* __restrict__ wq,
                           const float* __restrict__ scale,
                           float* __restrict__ out) {
  const int lane  = threadIdx.x & 31;
  const int wave  = threadIdx.x >> 5;
  const int lr    = lane & 15;       // row within half-wave
  const int half  = lane >> 4;       // 0 or 1

  const int n0     = blockIdx.x * 32;      // first 16-wide N tile
  const int m_base = wave * 16;            // 16-wide M tile

  const float* xrow  = x  + (m_base + lr) * IN_F;    // A: M = lr (both halves)
  const float* wrow0 = wq + (n0 + lr) * IN_F;        // B tile 0: N = lr
  const float* wrow1 = wrow0 + 16 * IN_F;            // B tile 1: N = lr + 16
  // n0 and n0+16 are always inside the same 128-wide scale block
  const float* srow  = scale + (n0 / BLK) * NKB;

  v8f acc0 = {};
  v8f acc1 = {};

  for (int kb = 0; kb < NKB; ++kb) {
    const int k0 = kb * BLK;
    // speculative prefetch of the next K-block of both weight tiles
    __builtin_prefetch(wrow0 + k0 + BLK, 0, 0);
    __builtin_prefetch(wrow1 + k0 + BLK, 0, 0);

    v8f c0 = {};
    v8f c1 = {};
#pragma unroll
    for (int kk = 0; kk < 4; ++kk) {
      const int kw = k0 + kk * 32;

      // A fragment: half 0 -> K {0..15}, half 1 -> K {16..31} (contiguous, matches B)
      const int ka = kw + half * 16;
      v4f a0 = *(const v4f*)(xrow + ka);
      v4f a1 = *(const v4f*)(xrow + ka + 4);
      v4f a2 = *(const v4f*)(xrow + ka + 8);
      v4f a3 = *(const v4f*)(xrow + ka + 12);
      v16bf A = cvt16(a0, a1, a2, a3);

      // B fragments: half 0 -> K {0..15}, half 1 -> K {16..31} (contiguous)
      const int kbf = kw + half * 16;
      v16bf B0 = loadB(wrow0 + kbf);
      v16bf B1 = loadB(wrow1 + kbf);

      // one A, two WMMAs -> A load/convert amortized over 2 N-tiles
      c0 = __builtin_amdgcn_wmma_f32_16x16x32_bf16(
          false, A, false, B0, (short)0, c0, false, false);
      c1 = __builtin_amdgcn_wmma_f32_16x16x32_bf16(
          false, A, false, B1, (short)0, c1, false, false);
    }

    // block-scale dequant folded into accumulation; one scalar serves both tiles
    const float s = srow[kb];
#pragma unroll
    for (int i = 0; i < 8; ++i) {
      acc0[i] = fmaf(c0[i], s, acc0[i]);
      acc1[i] = fmaf(c1[i], s, acc1[i]);
    }
  }

  // C/D layout: lane -> N ; VGPR r -> M = m_base + r + half*8
  float* ocol0 = out + n0 + lr;
  float* ocol1 = ocol0 + 16;
#pragma unroll
  for (int r = 0; r < 8; ++r) {
    const int m = m_base + r + half * 8;
    ocol0[m * OUT_F] = acc0[r];
    ocol1[m * OUT_F] = acc1[r];
  }
}

extern "C" void kernel_launch(void* const* d_in, const int* in_sizes, int n_in,
                              void* d_out, int out_size, void* d_ws, size_t ws_size,
                              hipStream_t stream) {
  (void)in_sizes; (void)n_in; (void)out_size; (void)d_ws; (void)ws_size;
  const float* x     = (const float*)d_in[0];   // [64, 7168]
  const float* wq    = (const float*)d_in[1];   // [18432, 7168]
  const float* scale = (const float*)d_in[2];   // [144, 56]
  float* out = (float*)d_out;                   // [64, 18432]

  dim3 grid(OUT_F / 32);   // 576 blocks, each owns 32 output features
  dim3 block(128);         // 4 waves, one M-tile each, 2 N-tiles per wave
  dequant_gemm_bf16wmma<<<grid, block, 0, stream>>>(x, wq, scale, out);
}
